// CrossmodalFusion_12558484374140
// MI455X (gfx1250) — compile-verified, matrix operand order checked by reading, not run on hardware
//
#include <hip/hip_runtime.h>
#include <math.h>

// ---------------- problem constants ----------------
#define Bsz   1024
#define Rr    36
#define Dd    1024
#define Hh    512
#define Cc    133
#define Mtot  (Bsz * Rr)   // 36864

// ---------------- WMMA types ----------------
typedef __attribute__((ext_vector_type(16))) __bf16 v16bf;
typedef __attribute__((ext_vector_type(8)))  float  v8f;
typedef __attribute__((ext_vector_type(4)))  int    v4i;

__device__ __forceinline__ unsigned short f32_to_bf16(float f) {
    union { float f; unsigned int u; } v; v.f = f;
    unsigned int u = v.u;
    unsigned int r = u + 0x7FFFu + ((u >> 16) & 1u);  // round-to-nearest-even
    return (unsigned short)(r >> 16);
}

// ---------------- async global->LDS support (gfx1250) ----------------
#if __has_builtin(__builtin_amdgcn_global_load_async_to_lds_b128) && \
    __has_builtin(__builtin_amdgcn_s_wait_asynccnt)
#define ASYNC_OK 1
#else
#define ASYNC_OK 0
#endif

#if ASYNC_OK
typedef __attribute__((address_space(1))) v4i gv4i;   // global v4i
typedef __attribute__((address_space(3))) v4i lv4i;   // LDS v4i
__device__ __forceinline__ void async_copy16(const void* g, void* l) {
    // global flat address == global address; flat LDS address truncates to LDS offset
    __builtin_amdgcn_global_load_async_to_lds_b128(
        (gv4i*)(unsigned long long)g,
        (lv4i*)(unsigned int)(unsigned long long)l,
        0, 0);
}
#endif

// ---------------- GEMM tile config ----------------
#define BM 128
#define BN 128
#define BK 32
#define LDSTR 40      // ushort stride: 80B rows -> 16B aligned, bank-spread

// Gather one 16(row)x32(K) bf16 fragment from LDS (row-major, stride LDSTR).
// 16-bit operand lane layout (ISA 7.12.2):
//   lanes 0-15 : row=lane,    K = 0..7  then 16..23
//   lanes16-31 : row=lane-16, K = 8..15 then 24..31
__device__ __forceinline__ v16bf load_frag(const unsigned short* lds,
                                           int rowBase, int r16, int half) {
    union { uint4 u[2]; v16bf v; } f;
    const unsigned short* p = lds + (rowBase + r16) * LDSTR + half * 8;
    f.u[0] = *(const uint4*)(p);
    f.u[1] = *(const uint4*)(p + 16);
    return f.v;
}

// C = A(MxK,bf16) @ Wt^T (Wt is NxK bf16) with fused epilogues.
// Block tile 128x128, 8 waves of 32x64 (2 M-sub x 4 N-sub), K step 32,
// double-buffered LDS staging via GLOBAL_LOAD_ASYNC_TO_LDS_B128 when available.
// MODE 0: outB = bf16(relu(acc + bias))                          (hidden pass)
// MODE 1: outF = acc + bias + resid                              (residual pass)
// MODE 2: s = tanh(acc+bias); outB = bf16(resid*s + mask?q2:0)   (scaling pass)
// MODE 3: outF = relu(acc + bias)                                (final pass)
template <int MODE>
__global__ __launch_bounds__(256) void gemm_wmma(
    const unsigned short* __restrict__ A,
    const unsigned short* __restrict__ Wt,
    const float* __restrict__ bias,
    float* __restrict__ outF,
    unsigned short* __restrict__ outB,
    const float* __restrict__ resid,
    const float* __restrict__ q2,
    const int*   __restrict__ lens,
    int M, int N, int K)
{
    __shared__ __align__(16) unsigned short ldsA[2][BM * LDSTR];
    __shared__ __align__(16) unsigned short ldsB[2][BN * LDSTR];

    const int tid  = threadIdx.x;
    const int wave = tid >> 5;
    const int lane = tid & 31;
    const int half = lane >> 4;
    const int r16  = lane & 15;
    const int wm   = wave >> 1;   // 0..3 : 32-row slice
    const int wn   = wave & 1;    // 0..1 : 64-col slice
    const int mBase = blockIdx.y * BM;
    const int nBase = blockIdx.x * BN;

    // stage one 128x32 A tile + 128x32 B tile into buffer `sel`
    auto stage = [&](int kt, int sel) {
#pragma unroll
        for (int j = 0; j < 2; ++j) {
            int c   = tid + 256 * j;          // 512 chunks of 8 bf16 per tile
            int row = c >> 2;
            int kc  = (c & 3) * 8;
            const unsigned short* gA = A  + (size_t)(mBase + row) * K + kt + kc;
            const unsigned short* gB = Wt + (size_t)(nBase + row) * K + kt + kc;
            unsigned short* lA = &ldsA[sel][row * LDSTR + kc];
            unsigned short* lB = &ldsB[sel][row * LDSTR + kc];
#if ASYNC_OK
            async_copy16(gA, lA);
            async_copy16(gB, lB);
#else
            *(uint4*)lA = *(const uint4*)gA;
            *(uint4*)lB = *(const uint4*)gB;
            __builtin_prefetch(gA + BK, 0, 1);
            __builtin_prefetch(gB + BK, 0, 1);
#endif
        }
    };

    v8f acc[2][4];
#pragma unroll
    for (int mi = 0; mi < 2; ++mi)
#pragma unroll
        for (int ni = 0; ni < 4; ++ni)
#pragma unroll
            for (int e = 0; e < 8; ++e) acc[mi][ni][e] = 0.f;

    stage(0, 0);
#if ASYNC_OK
    __builtin_amdgcn_s_wait_asynccnt(0);
#endif
    __syncthreads();

    int buf = 0;
    for (int kt = 0; kt < K; kt += BK) {
        if (kt + BK < K) stage(kt + BK, buf ^ 1);   // overlap next tile with compute

        v16bf afrag[2], bfrag[4];
#pragma unroll
        for (int mi = 0; mi < 2; ++mi)
            afrag[mi] = load_frag(ldsA[buf], wm * 32 + mi * 16, r16, half);
#pragma unroll
        for (int ni = 0; ni < 4; ++ni)
            bfrag[ni] = load_frag(ldsB[buf], wn * 64 + ni * 16, r16, half);

#pragma unroll
        for (int mi = 0; mi < 2; ++mi)
#pragma unroll
            for (int ni = 0; ni < 4; ++ni)
                acc[mi][ni] = __builtin_amdgcn_wmma_f32_16x16x32_bf16(
                    false, afrag[mi], false, bfrag[ni],
                    (short)0, acc[mi][ni], false, false);

#if ASYNC_OK
        __builtin_amdgcn_s_wait_asynccnt(0);
#endif
        __syncthreads();
        buf ^= 1;
    }

    // epilogue; C/D layout: lane 0-15 -> N=lane, M=v ; lane 16-31 -> N=lane-16, M=v+8
#pragma unroll
    for (int mi = 0; mi < 2; ++mi)
#pragma unroll
        for (int ni = 0; ni < 4; ++ni)
#pragma unroll
            for (int v = 0; v < 8; ++v) {
                int row = mBase + wm * 32 + mi * 16 + half * 8 + v;
                int col = nBase + wn * 64 + ni * 16 + r16;
                size_t idx = (size_t)row * N + col;
                float a = acc[mi][ni][v] + bias[col];
                if constexpr (MODE == 0) {
                    a = a > 0.f ? a : 0.f;
                    outB[idx] = f32_to_bf16(a);
                } else if constexpr (MODE == 1) {
                    outF[idx] = a + resid[idx];
                } else if constexpr (MODE == 2) {
                    float s = tanhf(a);
                    int b  = row / Rr;
                    int rr = row - b * Rr;
                    float seg = (rr < lens[b]) ? q2[(size_t)b * N + col] : 0.f;
                    outB[idx] = f32_to_bf16(resid[idx] * s + seg);
                } else {
                    outF[idx] = a > 0.f ? a : 0.f;
                }
            }
}

// ---------------- small kernels ----------------

// f32 -> bf16 elementwise
__global__ void convert_bf16_kernel(const float* __restrict__ src,
                                    unsigned short* __restrict__ dst, size_t n) {
    size_t i = (size_t)blockIdx.x * 256 + threadIdx.x;
    if (i < n) dst[i] = f32_to_bf16(src[i]);
}

// W (KxN f32) -> Wt (NxK bf16)
__global__ void transpose_bf16_kernel(const float* __restrict__ W,
                                      unsigned short* __restrict__ Wt,
                                      int K, int N) {
    size_t i = (size_t)blockIdx.x * 256 + threadIdx.x;
    if (i >= (size_t)K * N) return;
    int k = (int)(i / N);
    int n = (int)(i - (size_t)k * N);
    Wt[(size_t)n * K + k] = f32_to_bf16(W[i]);
}

// mean over 7x7 spatial for each (b, c)
__global__ void pool_kernel(const float* __restrict__ segs,
                            float* __restrict__ means) {
    int b = blockIdx.x;
    for (int c = threadIdx.x; c < Cc; c += blockDim.x) {
        const float* p = segs + ((size_t)b * Cc + c) * 49;
        float s = 0.f;
#pragma unroll
        for (int i = 0; i < 49; ++i) s += p[i];
        means[(size_t)b * Cc + c] = s * (1.f / 49.f);
    }
}

// q_pre = means @ seg_W + seg_b   (small K=133, naive per-element)
__global__ void seg_gemm_kernel(const float* __restrict__ means,
                                const float* __restrict__ segW,
                                const float* __restrict__ segb,
                                float* __restrict__ qpre) {
    int b = blockIdx.y;
    int n = blockIdx.x * blockDim.x + threadIdx.x;
    float s = segb[n];
    for (int c = 0; c < Cc; ++c)
        s += means[(size_t)b * Cc + c] * segW[(size_t)c * Dd + n];
    qpre[(size_t)b * Dd + n] = s;
}

// q = layernorm(relu(q_pre)) * g + beta ; write f32 + bf16
__global__ __launch_bounds__(256) void relu_ln_kernel(
    const float* __restrict__ qpre,
    const float* __restrict__ g, const float* __restrict__ beta,
    float* __restrict__ qF, unsigned short* __restrict__ qB) {
    int b = blockIdx.x;
    int tid = threadIdx.x;
    __shared__ float red[256];
    __shared__ float sMean, sInv;

    float s = 0.f;
    for (int i = tid; i < Dd; i += 256) {
        float x = qpre[(size_t)b * Dd + i];
        s += (x > 0.f ? x : 0.f);
    }
    red[tid] = s; __syncthreads();
    for (int st = 128; st > 0; st >>= 1) {
        if (tid < st) red[tid] += red[tid + st];
        __syncthreads();
    }
    if (tid == 0) sMean = red[0] * (1.f / Dd);
    __syncthreads();
    float mean = sMean;

    float v = 0.f;
    for (int i = tid; i < Dd; i += 256) {
        float x = qpre[(size_t)b * Dd + i];
        x = x > 0.f ? x : 0.f;
        float d = x - mean;
        v += d * d;
    }
    __syncthreads();
    red[tid] = v; __syncthreads();
    for (int st = 128; st > 0; st >>= 1) {
        if (tid < st) red[tid] += red[tid + st];
        __syncthreads();
    }
    if (tid == 0) sInv = rsqrtf(red[0] * (1.f / Dd) + 1e-5f);
    __syncthreads();
    float inv = sInv;

    for (int i = tid; i < Dd; i += 256) {
        float x = qpre[(size_t)b * Dd + i];
        x = x > 0.f ? x : 0.f;
        float y = (x - mean) * inv * g[i] + beta[i];
        qF[(size_t)b * Dd + i] = y;
        qB[(size_t)b * Dd + i] = f32_to_bf16(y);
    }
}

// attn = dot(r, q2)/sqrt(D), masked sigmoid -> w ; overwrite r with wc = w*r
__global__ __launch_bounds__(128) void attn_wc_kernel(
    float* __restrict__ rwc,              // in: r (MxD), out: wc (in place)
    unsigned short* __restrict__ wcB,     // bf16 copy of wc
    const float* __restrict__ q2,         // BxD
    const int* __restrict__ lens) {
    int m  = blockIdx.x;
    int b  = m / Rr;
    int rr = m - b * Rr;
    int tid = threadIdx.x;
    const float* rrow = rwc + (size_t)m * Dd;
    const float* qrow = q2 + (size_t)b * Dd;

    float p = 0.f;
    for (int i = tid; i < Dd; i += 128) p += rrow[i] * qrow[i];
    __shared__ float red[128];
    red[tid] = p; __syncthreads();
    for (int st = 64; st > 0; st >>= 1) {
        if (tid < st) red[tid] += red[tid + st];
        __syncthreads();
    }
    float attn = red[0] * (1.f / 32.f);          // 1/sqrt(1024)
    float w = (rr < lens[b]) ? (1.f / (1.f + expf(-attn))) : 0.f;

    for (int i = tid; i < Dd; i += 128) {
        float wc = w * rrow[i];
        rwc[(size_t)m * Dd + i] = wc;
        wcB[(size_t)m * Dd + i] = f32_to_bf16(wc);
    }
}

// ---------------- launch ----------------
extern "C" void kernel_launch(void* const* d_in, const int* in_sizes, int n_in,
                              void* d_out, int out_size, void* d_ws, size_t ws_size,
                              hipStream_t stream) {
    const float* rgns   = (const float*)d_in[0];
    const float* segs   = (const float*)d_in[1];
    const int*   lens   = (const int*)d_in[2];
    const float* mi_W1  = (const float*)d_in[3];
    const float* mi_b1  = (const float*)d_in[4];
    const float* mi_W2  = (const float*)d_in[5];
    const float* mi_b2  = (const float*)d_in[6];
    const float* ms_W1  = (const float*)d_in[7];
    const float* ms_b1  = (const float*)d_in[8];
    const float* ms_W2  = (const float*)d_in[9];
    const float* ms_b2  = (const float*)d_in[10];
    const float* seg_W  = (const float*)d_in[11];
    const float* seg_b  = (const float*)d_in[12];
    const float* ln_g   = (const float*)d_in[13];
    const float* ln_b   = (const float*)d_in[14];
    const float* sc_W   = (const float*)d_in[15];
    const float* sc_b   = (const float*)d_in[16];
    const float* f1_W   = (const float*)d_in[17];
    const float* f1_b   = (const float*)d_in[18];
    float* out = (float*)d_out;

    // ---- workspace bump allocation (256B aligned) ----
    char* ws = (char*)d_ws;
    size_t off = 0;
    auto alloc = [&](size_t bytes) {
        size_t o = off;
        off = (off + bytes + 255) & ~(size_t)255;
        return (void*)(ws + o);
    };
    unsigned short* rgnsBf  = (unsigned short*)alloc((size_t)Mtot * Dd * 2); // aliased as wcBf later
    unsigned short* hidBf   = (unsigned short*)alloc((size_t)Mtot * Hh * 2);
    float*          rwcF    = (float*)alloc((size_t)Mtot * Dd * 4);          // r then wc (in place)
    unsigned short* zBf     = (unsigned short*)alloc((size_t)Mtot * Dd * 2);
    unsigned short* miW1t   = (unsigned short*)alloc((size_t)Hh * Dd * 2);
    unsigned short* miW2t   = (unsigned short*)alloc((size_t)Dd * Hh * 2);
    unsigned short* msW1t   = (unsigned short*)alloc((size_t)Hh * Dd * 2);
    unsigned short* msW2t   = (unsigned short*)alloc((size_t)Dd * Hh * 2);
    unsigned short* scWt    = (unsigned short*)alloc((size_t)Dd * Dd * 2);
    unsigned short* f1Wt    = (unsigned short*)alloc((size_t)Dd * Dd * 2);
    float*          means   = (float*)alloc((size_t)Bsz * Cc * 4);
    float*          qpre    = (float*)alloc((size_t)Bsz * Dd * 4);
    float*          qF      = (float*)alloc((size_t)Bsz * Dd * 4);
    unsigned short* qB      = (unsigned short*)alloc((size_t)Bsz * Dd * 2);
    unsigned short* mshBf   = (unsigned short*)alloc((size_t)Bsz * Hh * 2);
    float*          q2F     = (float*)alloc((size_t)Bsz * Dd * 4);
    unsigned short* wcBf    = rgnsBf;  // alias: rgns_bf16 dead after mi-hidden pass

    // ---- weight transpose+convert: W(KxN) f32 -> Wt(NxK) bf16 ----
    auto tgrid = [](int K, int N) { return dim3(((size_t)K * N + 255) / 256); };
    transpose_bf16_kernel<<<tgrid(Dd, Hh), 256, 0, stream>>>(mi_W1, miW1t, Dd, Hh);
    transpose_bf16_kernel<<<tgrid(Hh, Dd), 256, 0, stream>>>(mi_W2, miW2t, Hh, Dd);
    transpose_bf16_kernel<<<tgrid(Dd, Hh), 256, 0, stream>>>(ms_W1, msW1t, Dd, Hh);
    transpose_bf16_kernel<<<tgrid(Hh, Dd), 256, 0, stream>>>(ms_W2, msW2t, Hh, Dd);
    transpose_bf16_kernel<<<tgrid(Dd, Dd), 256, 0, stream>>>(sc_W,  scWt,  Dd, Dd);
    transpose_bf16_kernel<<<tgrid(Dd, Dd), 256, 0, stream>>>(f1_W,  f1Wt,  Dd, Dd);

    // ---- rgns -> bf16 ----
    {
        size_t n = (size_t)Mtot * Dd;
        convert_bf16_kernel<<<dim3((n + 255) / 256), 256, 0, stream>>>(rgns, rgnsBf, n);
    }

    // ---- q path ----
    pool_kernel<<<Bsz, 256, 0, stream>>>(segs, means);
    seg_gemm_kernel<<<dim3(Dd / 256, Bsz), 256, 0, stream>>>(means, seg_W, seg_b, qpre);
    relu_ln_kernel<<<Bsz, 256, 0, stream>>>(qpre, ln_g, ln_b, qF, qB);

    // ms MLP: msh = relu(q @ ms_W1 + b1); q2 = msh @ ms_W2 + b2 + q
    gemm_wmma<0><<<dim3(Hh / BN, Bsz / BM), 256, 0, stream>>>(
        qB, msW1t, ms_b1, nullptr, mshBf, nullptr, nullptr, nullptr, Bsz, Hh, Dd);
    gemm_wmma<1><<<dim3(Dd / BN, Bsz / BM), 256, 0, stream>>>(
        mshBf, msW2t, ms_b2, q2F, nullptr, qF, nullptr, nullptr, Bsz, Dd, Hh);

    // ---- r path (big): hid = relu(rgns @ mi_W1 + b1); r = hid @ mi_W2 + b2 + rgns
    gemm_wmma<0><<<dim3(Hh / BN, Mtot / BM), 256, 0, stream>>>(
        rgnsBf, miW1t, mi_b1, nullptr, hidBf, nullptr, nullptr, nullptr, Mtot, Hh, Dd);
    gemm_wmma<1><<<dim3(Dd / BN, Mtot / BM), 256, 0, stream>>>(
        hidBf, miW2t, mi_b2, rwcF, nullptr, rgns, nullptr, nullptr, Mtot, Dd, Hh);

    // ---- attention: r -> wc (in place f32) + bf16 copy ----
    attn_wc_kernel<<<Mtot, 128, 0, stream>>>(rwcF, wcBf, q2F, lens);

    // ---- z = wc * tanh(wc @ sc_W + sc_b) + masked seg_rep -> bf16
    gemm_wmma<2><<<dim3(Dd / BN, Mtot / BM), 256, 0, stream>>>(
        wcBf, scWt, sc_b, nullptr, zBf, rwcF, q2F, lens, Mtot, Dd, Dd);

    // ---- out = relu(z @ f1_W + f1_b)
    gemm_wmma<3><<<dim3(Dd / BN, Mtot / BM), 256, 0, stream>>>(
        zBf, f1Wt, f1_b, out, nullptr, nullptr, nullptr, nullptr, Mtot, Dd, Dd);

    (void)in_sizes; (void)n_in; (void)out_size; (void)ws_size;
}
